// LHIC_RNN_spectral_12790412607958
// MI455X (gfx1250) — matrix operand order repaired and checked by reading, not compiled, not run on hardware
//
#include <hip/hip_runtime.h>
#include <hip/hip_bf16.h>
#include <cstddef>
#include <cstdint>

// ---------------------------------------------------------------------------
// MI455X (gfx1250): GEMMs via v_wmma_f32_16x16x32_bf16 (wave32), weight panel
// staged into LDS by the Tensor Data Mover (tensor_load_to_lds + TENSORcnt).
// ---------------------------------------------------------------------------

typedef __attribute__((ext_vector_type(16))) __bf16       v16bf;
typedef __attribute__((ext_vector_type(8)))  float        v8f;
typedef __attribute__((ext_vector_type(4)))  unsigned int v4u;
typedef __attribute__((ext_vector_type(8)))  unsigned int v8u;

#define F_DIM 256
#define FH    1024
#define HH    160
#define WW    160
#define NPIX  (HH * WW)       // 25600
#define NLAY  4

#define TM 128   // GEMM block rows (8 waves x 16)
#define TN 16    // GEMM block cols (one WMMA tile wide)
#define TK 32    // K step (bf16 WMMA K)

__device__ __forceinline__ float wave_sum(float v) {
#pragma unroll
  for (int off = 16; off > 0; off >>= 1) v += __shfl_xor(v, off, 32);
  return v;
}

__device__ __forceinline__ float sigm(float x) { return 1.f / (1.f + __expf(-x)); }

// --------------------------- conv stem -------------------------------------
__global__ __launch_bounds__(256) void k_stem(
    const float* __restrict__ x, const float* __restrict__ w1,
    const float* __restrict__ b1, const float* __restrict__ w2,
    const float* __restrict__ b2, float* __restrict__ d) {
  int f = threadIdx.x;
  int n = blockIdx.x;
  int h = n / WW, w = n % WW;
  float c1 = w1[f], c0 = b1[f];
  float acc = 0.f;
#pragma unroll
  for (int ky = 0; ky < 7; ++ky) {
    int hy = h + ky - 3;
    if (hy < 0 || hy >= HH) continue;
#pragma unroll
    for (int kx = 0; kx < 7; ++kx) {
      int wx = w + kx - 3;
      if (wx < 0 || wx >= WW) continue;
      float t = x[hy * WW + wx] * c1 + c0;
      acc += w2[f * 49 + ky * 7 + kx] * t;
    }
  }
  acc += b2[f];
  d[(size_t)n * F_DIM + f] = acc * sigm(4.f * acc);   // WSiLU
}

// --------------------------- in-place LayerNorm (ln0) ----------------------
__global__ __launch_bounds__(256) void k_ln(
    float* __restrict__ d, const float* __restrict__ lw,
    const float* __restrict__ lb) {
  int wave = threadIdx.x >> 5, lane = threadIdx.x & 31;
  size_t row = (size_t)blockIdx.x * 8 + wave;
  float* dr = d + row * F_DIM;
  float v[8], s = 0.f;
#pragma unroll
  for (int j = 0; j < 8; ++j) { v[j] = dr[lane + j * 32]; s += v[j]; }
  s = wave_sum(s);
  float mu = s * (1.f / F_DIM), q = 0.f;
#pragma unroll
  for (int j = 0; j < 8; ++j) { float t = v[j] - mu; q += t * t; }
  q = wave_sum(q);
  float rs = rsqrtf(q * (1.f / F_DIM) + 1e-5f);
#pragma unroll
  for (int j = 0; j < 8; ++j) {
    int f = lane + j * 32;
    dr[f] = (v[j] - mu) * rs * lw[f] + lb[f];
  }
}

// ------------------- fused LN + token-shift mixing (SA) --------------------
__global__ __launch_bounds__(256) void k_sa_mix(
    const float* __restrict__ d, const float* __restrict__ lw,
    const float* __restrict__ lb, const float* __restrict__ sp,
    const float* __restrict__ tmk, const float* __restrict__ tmv,
    const float* __restrict__ tmr, __bf16* __restrict__ xk,
    __bf16* __restrict__ xv, __bf16* __restrict__ xr) {
  int wave = threadIdx.x >> 5, lane = threadIdx.x & 31;
  size_t row = (size_t)blockIdx.x * 8 + wave;
  const float* dr = d + row * F_DIM;
  float v[8], s = 0.f;
#pragma unroll
  for (int j = 0; j < 8; ++j) { v[j] = dr[lane + j * 32]; s += v[j]; }
  s = wave_sum(s);
  float mu = s * (1.f / F_DIM), q = 0.f;
#pragma unroll
  for (int j = 0; j < 8; ++j) { float t = v[j] - mu; q += t * t; }
  q = wave_sum(q);
  float rs = rsqrtf(q * (1.f / F_DIM) + 1e-5f);
#pragma unroll
  for (int j = 0; j < 8; ++j) {
    int f = lane + j * 32;
    size_t idx = row * F_DIM + f;
    float xs = (v[j] - mu) * rs * lw[f] + lb[f];
    float spv = sp[idx];
    xk[idx] = (__bf16)(xs * tmk[f] + spv * (1.f - tmk[f]));
    xv[idx] = (__bf16)(xs * tmv[f] + spv * (1.f - tmv[f]));
    xr[idx] = (__bf16)(xs * tmr[f] + spv * (1.f - tmr[f]));
  }
}

// ------------------- fused LN + token-shift mixing (FFN) -------------------
__global__ __launch_bounds__(256) void k_ffn_mix(
    const float* __restrict__ d, const float* __restrict__ lw,
    const float* __restrict__ lb, const float* __restrict__ sp,
    const float* __restrict__ tmk, const float* __restrict__ tmr,
    __bf16* __restrict__ xk, __bf16* __restrict__ xr) {
  int wave = threadIdx.x >> 5, lane = threadIdx.x & 31;
  size_t row = (size_t)blockIdx.x * 8 + wave;
  const float* dr = d + row * F_DIM;
  float v[8], s = 0.f;
#pragma unroll
  for (int j = 0; j < 8; ++j) { v[j] = dr[lane + j * 32]; s += v[j]; }
  s = wave_sum(s);
  float mu = s * (1.f / F_DIM), q = 0.f;
#pragma unroll
  for (int j = 0; j < 8; ++j) { float t = v[j] - mu; q += t * t; }
  q = wave_sum(q);
  float rs = rsqrtf(q * (1.f / F_DIM) + 1e-5f);
#pragma unroll
  for (int j = 0; j < 8; ++j) {
    int f = lane + j * 32;
    size_t idx = row * F_DIM + f;
    float xs = (v[j] - mu) * rs * lw[f] + lb[f];
    float spv = sp[idx];
    xk[idx] = (__bf16)(xs * tmk[f] + spv * (1.f - tmk[f]));
    xr[idx] = (__bf16)(xs * tmr[f] + spv * (1.f - tmr[f]));
  }
}

// ----------------- weight convert: f32 -> bf16, layout preserved -----------
// (row-major [M,K] is exactly the CDNA5 B-fragment feed: one output column
//  per lane = one weight row, 8 consecutive K per 16B chunk)
__global__ __launch_bounds__(256) void k_wconv(
    const float* __restrict__ W, __bf16* __restrict__ Wb, int count) {
  int i = blockIdx.x * 256 + threadIdx.x;
  if (i < count) Wb[i] = (__bf16)W[i];
}

// --------------------------- WMMA GEMM -------------------------------------
// C[N,M] (+)= A[N,K](bf16 row-major) x W[M,K](bf16 row-major, used as B^T)
// Weight panel (16 rows x K) is DMA'd into LDS once per block by the TDM.
// mode 0: C = acc + bias   1: C += acc + bias   2: C += gate*acc
// mode 3: C = sigmoid(acc) 4: C = relu(acc)^2 (bf16 out)
__global__ __launch_bounds__(256) void k_gemm(
    const __bf16* __restrict__ A, const __bf16* __restrict__ W,
    const float* __restrict__ bias, const float* __restrict__ gate,
    void* __restrict__ Cv, int Nrows, int K, int M, int mode) {
  extern __shared__ __bf16 Bs[];   // dynamic LDS only -> descriptor lds_addr = 0
  int tid = threadIdx.x;
  int wave = tid >> 5, lane = tid & 31;
  int n0 = blockIdx.x * TM;
  int m0 = blockIdx.y * TN;

  if (wave == 0) {
    // --- Tensor Data Mover: 2D tile (K x 16 rows, 2B elems) global -> LDS ---
    uint64_t ga = (uint64_t)(uintptr_t)(W + (size_t)m0 * K);
    unsigned uK = (unsigned)K;
    v4u g0;
    g0[0] = 1u;                                     // count=1, user D#
    g0[1] = 0u;                                     // lds_addr = 0
    g0[2] = (unsigned)ga;                           // global_addr[31:0]
    g0[3] = (unsigned)((ga >> 32) & 0x01ffffffu) | (2u << 30);  // addr[56:32], type=2
    v8u g1;
    g1[0] = 1u << 16;                               // data_size = 2 bytes
    g1[1] = (uK & 0xffffu) << 16;                   // tensor_dim0[15:0]
    g1[2] = (uK >> 16) | (16u << 16);               // tensor_dim0[31:16], tensor_dim1=16
    g1[3] = (uK & 0xffffu) << 16;                   // tile_dim0 = K
    g1[4] = 16u;                                    // tile_dim1 = 16, tile_dim2 = 0
    g1[5] = uK;                                     // tensor_dim0_stride[31:0]
    g1[6] = 0u;                                     // stride hi, dim1_stride lo
    g1[7] = 0u;
    v4u gz; gz[0] = 0u; gz[1] = 0u; gz[2] = 0u; gz[3] = 0u;
    asm volatile("tensor_load_to_lds %0, %1, %2, %3"
                 :: "s"(g0), "s"(g1), "s"(gz), "s"(gz) : "memory");
    __builtin_amdgcn_s_wait_tensorcnt(0);
  }
  __syncthreads();

  // A fragment (16x32 bf16): lane<16 -> K {0..7,16..23}; lane>=16 -> {8..15,24..31}
  const __bf16* Arow = A + (size_t)(n0 + wave * 16 + (lane & 15)) * K;
  // B fragment (32x16): lane = output col; lane<16 -> K {0..7,16..23}; lane>=16 -> {8..15,24..31}
  const __bf16* Brow = Bs + (size_t)(lane & 15) * K;
  int akb = (lane < 16) ? 0 : 8;
  int bkb = (lane < 16) ? 0 : 8;

  v8f acc = {};
#pragma unroll 4
  for (int k0 = 0; k0 < K; k0 += TK) {
    union { uint4 q[2]; v16bf v; } fa, fb;
    fa.q[0] = *(const uint4*)(Arow + k0 + akb);
    fa.q[1] = *(const uint4*)(Arow + k0 + akb + 16);
    fb.q[0] = *(const uint4*)(Brow + k0 + bkb);
    fb.q[1] = *(const uint4*)(Brow + k0 + bkb + 16);
    acc = __builtin_amdgcn_wmma_f32_16x16x32_bf16(
        false, fa.v, false, fb.v, (short)0, acc, false, false);
  }

  // epilogue: VGPR j holds row (lane<16 ? j : j+8), col = lane&15
  int col = m0 + (lane & 15);
  float b = bias ? bias[col] : 0.f;
  int rbase = n0 + wave * 16 + ((lane < 16) ? 0 : 8);
#pragma unroll
  for (int j = 0; j < 8; ++j) {
    size_t idx = (size_t)(rbase + j) * M + col;
    float v = acc[j] + b;
    if (mode == 0)       ((float*)Cv)[idx] = v;
    else if (mode == 1)  ((float*)Cv)[idx] += v;
    else if (mode == 2)  ((float*)Cv)[idx] += gate[idx] * v;
    else if (mode == 3)  ((float*)Cv)[idx] = sigm(v);
    else { float r = v > 0.f ? v : 0.f; ((__bf16*)Cv)[idx] = (__bf16)(r * r); }
  }
}

// ------------------- fused WKV (max-trick) + sigmoid(r) gate ---------------
__global__ __launch_bounds__(256) void k_wkv(
    const float* __restrict__ kk, const float* __restrict__ vv,
    const float* __restrict__ rp, const float* __restrict__ aa,
    const float* __restrict__ bb, const float* __restrict__ pp,
    const float* __restrict__ tf, __bf16* __restrict__ out) {
  size_t i = (size_t)blockIdx.x * 256 + threadIdx.x;
  int f = (int)(i & (F_DIM - 1));
  float ww = tf[f] + kk[i];
  float p  = fmaxf(pp[i], ww);
  float e1 = __expf(pp[i] - p), e2 = __expf(ww - p);
  float wkv = (e1 * aa[i] + e2 * vv[i]) / (e1 * bb[i] + e2);
  out[i] = (__bf16)(sigm(rp[i]) * wkv);
}

// -------------------- spatial depthwise 7x7 conv (HWC) ---------------------
__global__ __launch_bounds__(256) void k_dwconv(
    const float* __restrict__ d, const float* __restrict__ w,
    const float* __restrict__ bias, __bf16* __restrict__ out) {
  int f = threadIdx.x;
  int n = blockIdx.x;
  int h = n / WW, wq = n % WW;
  float acc = bias[f];
#pragma unroll
  for (int ky = 0; ky < 7; ++ky) {
    int hy = h + ky - 3;
    if (hy < 0 || hy >= HH) continue;
#pragma unroll
    for (int kx = 0; kx < 7; ++kx) {
      int wx = wq + kx - 3;
      if (wx < 0 || wx >= WW) continue;
      acc += w[f * 49 + ky * 7 + kx] * d[((size_t)hy * WW + wx) * F_DIM + f];
    }
  }
  out[(size_t)n * F_DIM + f] = (__bf16)acc;
}

// ------------------------- final HWC -> CHW transpose ----------------------
__global__ __launch_bounds__(256) void k_out(
    const float* __restrict__ d, float* __restrict__ out) {
  int f = threadIdx.x;
  int n = blockIdx.x;
  out[(size_t)f * NPIX + n] = d[(size_t)n * F_DIM + f];
}

// ===========================================================================
extern "C" void kernel_launch(void* const* d_in, const int* in_sizes, int n_in,
                              void* d_out, int out_size, void* d_ws,
                              size_t ws_size, hipStream_t stream) {
  const float* x       = (const float*)d_in[0];
  const float* cin_w1  = (const float*)d_in[1];
  const float* cin_b1  = (const float*)d_in[2];
  const float* cin_w2  = (const float*)d_in[3];
  const float* cin_b2  = (const float*)d_in[4];
  const float* ln0_w   = (const float*)d_in[5];
  const float* ln0_b   = (const float*)d_in[6];
  const float* ln1_w   = (const float*)d_in[7];
  const float* ln1_b   = (const float*)d_in[8];
  const float* ln2_w   = (const float*)d_in[9];
  const float* ln2_b   = (const float*)d_in[10];
  const float* att_tmk = (const float*)d_in[11];
  const float* att_tmv = (const float*)d_in[12];
  const float* att_tmr = (const float*)d_in[13];
  const float* ffn_tmk = (const float*)d_in[14];
  const float* ffn_tmr = (const float*)d_in[15];
  const float* att_tf  = (const float*)d_in[16];
  /* att_td = d_in[17] unused: single-step WKV returns no state */
  const float* att_kw  = (const float*)d_in[18];
  const float* att_vw  = (const float*)d_in[19];
  const float* att_rw  = (const float*)d_in[20];
  const float* att_ow  = (const float*)d_in[21];
  const float* ffn_rw  = (const float*)d_in[22];
  const float* ffn_kw  = (const float*)d_in[23];
  const float* ffn_vw  = (const float*)d_in[24];
  const float* spa_dw_w = (const float*)d_in[25];
  const float* spa_dw_b = (const float*)d_in[26];
  const float* spa_pw_w = (const float*)d_in[27];
  const float* spa_pw_b = (const float*)d_in[28];
  const float* state0   = (const float*)d_in[29];

  const int N = NPIX, F = F_DIM;
  const size_t NF = (size_t)N * F;

  // workspace layout
  char* ws = (char*)d_ws;
  float*  dbuf  = (float*)(ws);                         // N*F f32
  __bf16* act0  = (__bf16*)(ws + NF * 4);               // N*F bf16
  __bf16* act1  = (__bf16*)(ws + NF * 4 + NF * 2);
  __bf16* act2  = (__bf16*)(ws + NF * 4 + NF * 4);
  __bf16* hbuf  = (__bf16*)(ws + NF * 4 + NF * 6);      // N*FH bf16
  float*  tmpf  = (float*)(ws + NF * 4 + NF * 6 + (size_t)N * FH * 2);
  float*  gate  = (float*)((char*)tmpf + NF * 12);      // N*F f32
  __bf16* wb    = (__bf16*)((char*)gate + NF * 4);      // up to 1024*256 bf16

  float* kbuf = tmpf;
  float* vbuf = tmpf + NF;
  float* rbuf = tmpf + 2 * NF;

  dim3 blk(256);
  dim3 gemmF(N / TM, F / TN);
  dim3 gemmH(N / TM, FH / TN);
  const size_t smF = (size_t)TN * F * 2;    // 8 KB weight panel
  const size_t smH = (size_t)TN * FH * 2;   // 32 KB weight panel

  k_stem<<<N, blk, 0, stream>>>(x, cin_w1, cin_b1, cin_w2, cin_b2, dbuf);
  k_ln<<<N / 8, blk, 0, stream>>>(dbuf, ln0_w, ln0_b);

  for (int i = 0; i < NLAY; ++i) {
    const float* sp0 = state0 + (size_t)(5 * i + 0) * NF;
    const float* sp1 = state0 + (size_t)(5 * i + 1) * NF;
    const float* aa  = state0 + (size_t)(5 * i + 2) * NF;
    const float* bb  = state0 + (size_t)(5 * i + 3) * NF;
    const float* pp  = state0 + (size_t)(5 * i + 4) * NF;

    // ---- SA ----
    k_sa_mix<<<N / 8, blk, 0, stream>>>(dbuf, ln1_w + i * F, ln1_b + i * F,
                                        sp1, att_tmk + i * F, att_tmv + i * F,
                                        att_tmr + i * F, act0, act1, act2);
    k_wconv<<<(F * F) / 256, blk, 0, stream>>>(att_kw + (size_t)i * F * F, wb, F * F);
    k_gemm<<<gemmF, blk, smF, stream>>>(act0, wb, nullptr, nullptr, kbuf, N, F, F, 0);
    k_wconv<<<(F * F) / 256, blk, 0, stream>>>(att_vw + (size_t)i * F * F, wb, F * F);
    k_gemm<<<gemmF, blk, smF, stream>>>(act1, wb, nullptr, nullptr, vbuf, N, F, F, 0);
    k_wconv<<<(F * F) / 256, blk, 0, stream>>>(att_rw + (size_t)i * F * F, wb, F * F);
    k_gemm<<<gemmF, blk, smF, stream>>>(act2, wb, nullptr, nullptr, rbuf, N, F, F, 0);
    k_wkv<<<(unsigned)(NF / 256), blk, 0, stream>>>(kbuf, vbuf, rbuf, aa, bb, pp,
                                                    att_tf + i * F, act2);
    k_wconv<<<(F * F) / 256, blk, 0, stream>>>(att_ow + (size_t)i * F * F, wb, F * F);
    k_gemm<<<gemmF, blk, smF, stream>>>(act2, wb, nullptr, nullptr, dbuf, N, F, F, 1);

    // ---- FFN ----
    k_ffn_mix<<<N / 8, blk, 0, stream>>>(dbuf, ln2_w + i * F, ln2_b + i * F,
                                         sp0, ffn_tmk + i * F, ffn_tmr + i * F,
                                         act0, act1);
    k_wconv<<<(F * F) / 256, blk, 0, stream>>>(ffn_rw + (size_t)i * F * F, wb, F * F);
    k_gemm<<<gemmF, blk, smF, stream>>>(act1, wb, nullptr, nullptr, gate, N, F, F, 3);
    k_wconv<<<(FH * F) / 256, blk, 0, stream>>>(ffn_kw + (size_t)i * FH * F, wb, FH * F);
    k_gemm<<<gemmH, blk, smF, stream>>>(act0, wb, nullptr, nullptr, hbuf, N, F, FH, 4);
    k_wconv<<<(FH * F) / 256, blk, 0, stream>>>(ffn_vw + (size_t)i * F * FH, wb, FH * F);
    k_gemm<<<gemmF, blk, smH, stream>>>(hbuf, wb, nullptr, gate, dbuf, N, FH, F, 2);

    // ---- spatial residual ----
    k_dwconv<<<N, blk, 0, stream>>>(dbuf, spa_dw_w + (size_t)i * F * 49,
                                    spa_dw_b + i * F, act0);
    k_wconv<<<(F * F) / 256, blk, 0, stream>>>(spa_pw_w + (size_t)i * F * F, wb, F * F);
    k_gemm<<<gemmF, blk, smF, stream>>>(act0, wb, spa_pw_b + i * F, nullptr, dbuf,
                                        N, F, F, 1);
  }

  k_out<<<N, blk, 0, stream>>>(dbuf, (float*)d_out);
}
